// UniPhyBlock_18657337934657
// MI455X (gfx1250) — matrix operand order repaired and checked by reading, not compiled, not run on hardware
//
#include <hip/hip_runtime.h>
#include <hip/hip_bf16.h>
#include <math.h>

// ---------------------------------------------------------------------------
// UniPhyBlock forward for MI455X (gfx1250, wave32, WMMA).
// GEMM stages (complex 3x3 conv, eigen encode/decode, MoE FFN) use
// v_wmma_f32_16x16x32_bf16.  All WMMA fragments are fed by 2x b128 vector
// loads: weights are pre-swizzled into fragment order (bf16), activations are
// stored bf16 K-contiguous by the producing kernel.
// ---------------------------------------------------------------------------

typedef __bf16 bf16_t;
typedef __attribute__((ext_vector_type(16))) __bf16 v16bf;
typedef __attribute__((ext_vector_type(8)))  __bf16 v8bf;
typedef __attribute__((ext_vector_type(8)))  float  v8f;

#define N_B 2
#define N_T 16
#define N_D 64
#define N_H 48
#define N_W 48
#define HW  (N_H * N_W)              // 2304
#define N_IMG (N_B * N_T)            // 32
#define N_POS (N_IMG * HW)           // 73728
#define IMG_STRIDE (N_D * HW)        // 147456
#define N_EXP 4
#define N_F 256
#define FRAG_ELEMS 512               // 32 lanes x 16 bf16 per fragment

// ------------------------------ helpers -----------------------------------

__device__ __forceinline__ v8f wmma_bf16(v16bf a, v16bf b, v8f c) {
  return __builtin_amdgcn_wmma_f32_16x16x32_bf16(
      /*neg_a=*/false, a, /*neg_b=*/false, b,
      /*c_mod=*/(short)0, c, /*reuse_a=*/false, /*reuse_b=*/false);
}

__device__ __forceinline__ v8f v8f_zero() {
  v8f z = {0.f, 0.f, 0.f, 0.f, 0.f, 0.f, 0.f, 0.f};
  return z;
}

__device__ __forceinline__ v16bf v16bf_zero() {
  v16bf z;
#pragma unroll
  for (int e = 0; e < 16; ++e) z[e] = (__bf16)0.f;
  return z;
}

__device__ __forceinline__ v16bf combine(v8bf lo, v8bf hi) {
  v16bf f;
#pragma unroll
  for (int e = 0; e < 8; ++e) {
    f[e] = lo[e];
    f[e + 8] = hi[e];
  }
  return f;
}

// A fragment (16x32 bf16) from a K-contiguous row.  Per ISA layout:
// lane<16: row M=lane, K {0..7} U {16..23}; lane>=16: same row, K {8..15} U {24..31}.
// Both halves are 8 consecutive bf16 -> one b128 load each.
__device__ __forceinline__ v16bf load_a_row(const bf16_t* row, int k0, int lane) {
  int half = (lane >> 4) & 1;
  const bf16_t* r = row + k0 + half * 8;
  v8bf lo = *(const v8bf*)r;
  v8bf hi = *(const v8bf*)(r + 16);
  return combine(lo, hi);
}

__device__ __forceinline__ v16bf load_a_rm(const bf16_t* base, int ld, int m, int k0, int lane) {
  return load_a_row(base + (size_t)m * ld, k0, lane);
}

// Pre-swizzled B fragment: 32 lanes x 16 bf16 contiguous -> 2x b128.
__device__ __forceinline__ v16bf load_frag(const bf16_t* fragbuf, int lane) {
  const v8bf* p = (const v8bf*)(fragbuf + (size_t)lane * 16);
  return combine(p[0], p[1]);
}

__device__ __forceinline__ float softplus_f(float x) {
  return (x > 20.f) ? x : log1pf(expf(x));
}

__device__ __forceinline__ float gelu_tanh(float x) {
  float x3 = x * x * x;
  float t = tanhf(0.7978845608028654f * (x + 0.044715f * x3));
  return 0.5f * x * (1.f + t);
}

// --------------------- weight pre-swizzle kernels --------------------------
// B-fragment gather (32x16 tile): lane<16 -> N=n0+lane, K {0..7} U {16..23};
// lane>=16 -> N=n0+lane-16, K {8..15} U {24..31}.  `scale`=+-1 implements B negation
// (bf16 WMMA has no A/B NEG modifier).

__global__ void swz_sq64_kernel(const float* __restrict__ src, bf16_t* __restrict__ dst,
                                float scale) {
  int w = threadIdx.x >> 5;  // 8 fragments: f = nt*2 + kc
  int lane = threadIdx.x & 31;
  int nt = w >> 1, kc = w & 1;
  int n = nt * 16 + (lane & 15);
  int k0 = kc * 32 + ((lane >> 4) & 1) * 8;
  bf16_t* o = dst + ((size_t)w * 32 + lane) * 16;
#pragma unroll
  for (int e = 0; e < 16; ++e)
    o[e] = (bf16_t)(scale * src[(size_t)(k0 + e + (e >= 8 ? 8 : 0)) * 64 + n]);
}

// conv weights (O,I,3,3): frag f = tap*8 + kc*4 + ct;  B[k=cin][n=cout]
__global__ void swz_conv_kernel(const float* __restrict__ w, bf16_t* __restrict__ dst,
                                float scale) {
  int gw = blockIdx.x * 8 + (threadIdx.x >> 5);
  if (gw >= 72) return;
  int lane = threadIdx.x & 31;
  int ct = gw & 3;
  int kc = (gw >> 2) & 1;
  int tap = gw >> 3;
  int co = ct * 16 + (lane & 15);
  int k0 = kc * 32 + ((lane >> 4) & 1) * 8;
  bf16_t* o = dst + ((size_t)gw * 32 + lane) * 16;
#pragma unroll
  for (int e = 0; e < 16; ++e)
    o[e] = (bf16_t)(scale * w[(size_t)co * 576 + (size_t)(k0 + e + (e >= 8 ? 8 : 0)) * 9 + tap]);
}

// W1 (4,128,256): frag f = e*64 + nt*4 + kc
__global__ void swz_w1_kernel(const float* __restrict__ W1, bf16_t* __restrict__ dst) {
  int gw = blockIdx.x * 8 + (threadIdx.x >> 5);  // 256 frags
  int lane = threadIdx.x & 31;
  int kc = gw & 3;
  int nt = (gw >> 2) & 15;
  int e2 = gw >> 6;
  int n = nt * 16 + (lane & 15);
  int k0 = kc * 32 + ((lane >> 4) & 1) * 8;
  const float* src = W1 + (size_t)e2 * 128 * N_F;
  bf16_t* o = dst + ((size_t)gw * 32 + lane) * 16;
#pragma unroll
  for (int e = 0; e < 16; ++e)
    o[e] = (bf16_t)src[(size_t)(k0 + e + (e >= 8 ? 8 : 0)) * N_F + n];
}

// W2 (4,256,128): frag f = e*64 + nt*8 + kc
__global__ void swz_w2_kernel(const float* __restrict__ W2, bf16_t* __restrict__ dst) {
  int gw = blockIdx.x * 8 + (threadIdx.x >> 5);  // 256 frags
  int lane = threadIdx.x & 31;
  int kc = gw & 7;
  int nt = (gw >> 3) & 7;
  int e2 = gw >> 6;
  int n = nt * 16 + (lane & 15);
  int k0 = kc * 32 + ((lane >> 4) & 1) * 8;
  const float* src = W2 + (size_t)e2 * N_F * 128;
  bf16_t* o = dst + ((size_t)gw * 32 + lane) * 16;
#pragma unroll
  for (int e = 0; e < 16; ++e)
    o[e] = (bf16_t)src[(size_t)(k0 + e + (e >= 8 ? 8 : 0)) * 128 + n];
}

// ------------------------- 1) spatial LayerNorm ----------------------------
// Output: position-major bf16 (pos*64 + cin), K-contiguous for conv A loads.

__global__ void ln1_kernel(const float* __restrict__ xre, const float* __restrict__ xim,
                           const float* __restrict__ gsg, const float* __restrict__ gsb,
                           bf16_t* __restrict__ nr, bf16_t* __restrict__ ni) {
  int p = blockIdx.x * blockDim.x + threadIdx.x;
  if (p >= N_POS) return;
  int img = p / HW, pix = p - img * HW;
  const float* br = xre + (size_t)img * IMG_STRIDE + pix;
  const float* bi = xim + (size_t)img * IMG_STRIDE + pix;
  float s = 0.f, ss = 0.f;
  for (int d = 0; d < N_D; ++d) {
    float a = br[(size_t)d * HW], c = bi[(size_t)d * HW];
    s += a + c;
    ss += a * a + c * c;
  }
  float mean = s * (1.f / 128.f);
  float var = ss * (1.f / 128.f) - mean * mean;
  float inv = rsqrtf(var + 1e-5f);
  bf16_t* orp = nr + (size_t)p * 64;
  bf16_t* oip = ni + (size_t)p * 64;
  for (int d = 0; d < N_D; ++d) {
    orp[d] = (bf16_t)((br[(size_t)d * HW] - mean) * inv * gsg[d] + gsb[d]);
    oip[d] = (bf16_t)((bi[(size_t)d * HW] - mean) * inv * gsg[64 + d] + gsb[64 + d]);
  }
}

// --------------- 2) complex 3x3 conv as implicit WMMA GEMM -----------------
// One wave = (img, y, 16-pixel x-tile, 16-cout tile).  Writes f32 planar
// (for final residual) and bf16 position-major (encode-GEMM A operand).

__global__ void __launch_bounds__(128) conv_wmma_kernel(
    const bf16_t* __restrict__ nr, const bf16_t* __restrict__ ni,
    const bf16_t* __restrict__ fWR, const bf16_t* __restrict__ fWI,
    const bf16_t* __restrict__ fWIN,
    const float* __restrict__ bcr, const float* __restrict__ bci,
    const float* __restrict__ metric,
    const float* __restrict__ xre_in, const float* __restrict__ xim_in,
    float* __restrict__ xpR, float* __restrict__ xpI,
    bf16_t* __restrict__ xtR, bf16_t* __restrict__ xtI) {
  int wid = blockIdx.x * (blockDim.x >> 5) + (threadIdx.x >> 5);
  int lane = threadIdx.x & 31;
  int ct = wid & 3;
  int xt = (wid >> 2) % 3;
  int y = (wid / 12) % N_H;
  int img = wid / (12 * N_H);
  if (img >= N_IMG) return;
  int cout0 = ct * 16;
  int x0 = xt * 16;
  int m = lane & 15;
  int half = (lane >> 4) & 1;

  v8f accR = v8f_zero(), accI = v8f_zero();
  for (int dy = -1; dy <= 1; ++dy) {
    int yc = y + dy;
    bool rowok = (yc >= 0) && (yc < N_H);
    int ycc = rowok ? yc : 0;
    for (int dx = -1; dx <= 1; ++dx) {
      int tap = (dy + 1) * 3 + (dx + 1);
      int xc = x0 + m + dx;  // this lane's A-row pixel
      bool ok = rowok && (xc >= 0) && (xc < N_W);
      int xcc = ok ? xc : 0;
      size_t rowoff = ((size_t)img * HW + (size_t)ycc * N_W + xcc) * 64;
#pragma unroll
      for (int kc = 0; kc < 2; ++kc) {
        int cin0 = kc * 32;
        v16bf ar = ok ? load_a_row(nr + rowoff, cin0, lane) : v16bf_zero();
        v16bf ai = ok ? load_a_row(ni + rowoff, cin0, lane) : v16bf_zero();
        size_t f = (size_t)(tap * 8 + kc * 4 + ct) * FRAG_ELEMS;
        v16bf brr = load_frag(fWR + f, lane);
        v16bf bin = load_frag(fWIN + f, lane);
        v16bf bip = load_frag(fWI + f, lane);
        accR = wmma_bf16(ar, brr, accR);
        accR = wmma_bf16(ai, bin, accR);
        accI = wmma_bf16(ar, bip, accI);
        accI = wmma_bf16(ai, brr, accI);
      }
    }
  }
  // C/D layout: elem v -> row M = v + 8*half (pixel), col N = lane&15 (cout)
  int n = lane & 15;
  const size_t imb = (size_t)img * IMG_STRIDE;
#pragma unroll
  for (int v = 0; v < 8; ++v) {
    int row = v + 8 * half;
    int px = x0 + row;
    int co = cout0 + n;
    float met = metric[y * N_W + px];
    size_t idx = imb + (size_t)co * HW + (size_t)y * N_W + px;
    float vr = xre_in[idx] + accR[v] * met + bcr[co];
    float vi = xim_in[idx] + accI[v] * met + bci[co];
    xpR[idx] = vr;
    xpI[idx] = vi;
    size_t pos = (size_t)img * HW + (size_t)y * N_W + px;
    xtR[pos * 64 + co] = (bf16_t)vr;
    xtI[pos * 64 + co] = (bf16_t)vi;
  }
}

// ---------------- 3/7) batched complex GEMM (M=73728, 64x64) ---------------
// A: bf16 position-major (pos*64+k).  B: pre-swizzled fragments (re, im, -im).
// C: f32 position-major.

__global__ void __launch_bounds__(128) cgemm64_kernel(
    const bf16_t* __restrict__ AR, const bf16_t* __restrict__ AI,
    const bf16_t* __restrict__ fBR, const bf16_t* __restrict__ fBI,
    const bf16_t* __restrict__ fBIN,
    float* __restrict__ CR, float* __restrict__ CI) {
  int wid = blockIdx.x * (blockDim.x >> 5) + (threadIdx.x >> 5);
  int lane = threadIdx.x & 31;
  int nt = wid & 3;
  int pt = wid >> 2;
  if (pt >= N_POS / 16) return;
  int n0 = nt * 16;
  size_t pos0 = (size_t)pt * 16;
  const bf16_t* arb = AR + pos0 * 64;
  const bf16_t* aib = AI + pos0 * 64;
  int m = lane & 15, half = (lane >> 4) & 1;
  v8f accR = v8f_zero(), accI = v8f_zero();
#pragma unroll
  for (int kc = 0; kc < 2; ++kc) {
    v16bf ar = load_a_rm(arb, 64, m, kc * 32, lane);
    v16bf ai = load_a_rm(aib, 64, m, kc * 32, lane);
    size_t f = (size_t)(nt * 2 + kc) * FRAG_ELEMS;
    v16bf brr = load_frag(fBR + f, lane);
    v16bf bip = load_frag(fBI + f, lane);
    v16bf bin = load_frag(fBIN + f, lane);
    accR = wmma_bf16(ar, brr, accR);
    accR = wmma_bf16(ai, bin, accR);
    accI = wmma_bf16(ar, bip, accI);
    accI = wmma_bf16(ai, brr, accI);
  }
#pragma unroll
  for (int v = 0; v < 8; ++v) {
    size_t o = (pos0 + v + 8 * half) * 64 + n0 + m;
    CR[o] = accR[v];
    CI[o] = accI[v];
  }
}

// ------------------------- 4) spatial mean of x_eigen ----------------------

__global__ void mean_kernel(const float* __restrict__ xeR, const float* __restrict__ xeI,
                            float* __restrict__ mR, float* __restrict__ mI) {
  int img = blockIdx.x;
  int e = threadIdx.x;
  const float* pr = xeR + (size_t)img * IMG_STRIDE + e;
  const float* pi = xeI + (size_t)img * IMG_STRIDE + e;
  float sr = 0.f, si = 0.f;
  for (int pix = 0; pix < HW; ++pix) {
    sr += pr[(size_t)pix * 64];
    si += pi[(size_t)pix * 64];
  }
  mR[img * 64 + e] = sr * (1.f / HW);
  mI[img * 64 + e] = si * (1.f / HW);
}

// ------------ 5) flux scan + source / gate / op_decay / op_forcing ---------

__global__ void flux_kernel(
    const float* __restrict__ mR, const float* __restrict__ mI,
    const float* __restrict__ dt, const float* __restrict__ lam_flux,
    const float* __restrict__ injR, const float* __restrict__ injI,
    const float* __restrict__ f0R, const float* __restrict__ f0I,
    const float* __restrict__ WsR, const float* __restrict__ WsI,
    const float* __restrict__ Wgate, const float* __restrict__ bgate,
    const float* __restrict__ lamRe, const float* __restrict__ lamIm,
    float* __restrict__ srcR, float* __restrict__ srcI,
    float* __restrict__ gbuf,
    float* __restrict__ decR, float* __restrict__ decI,
    float* __restrict__ forR, float* __restrict__ forI) {
  __shared__ float fshR[N_B][N_T][N_D];
  __shared__ float fshI[N_B][N_T][N_D];
  int tid = threadIdx.x;
  int b = tid >> 6, e = tid & 63;
  if (tid < N_B * N_D) {
    float sp = softplus_f(lam_flux[e]);
    float ir = injR[e], ii = injI[e];
    float sr = 0.f, si = 0.f, cum = 1.f;
    float fr0 = f0R[b * 64 + e], fi0 = f0I[b * 64 + e];
    for (int t = 0; t < N_T; ++t) {
      float a = expf(-sp * dt[t]);  // DT_REF = 1
      float xr = mR[(b * N_T + t) * 64 + e], xi = mI[(b * N_T + t) * 64 + e];
      float Xr = xr * ir - xi * ii, Xi = xr * ii + xi * ir;
      sr = a * sr + Xr;
      si = a * si + Xi;
      cum *= a;
      fshR[b][t][e] = sr + fr0 * cum;
      fshI[b][t][e] = si + fi0 * cum;
    }
    // op_decay = exp(lam*dt), op_forcing = (op_decay-1)/lam, lam = -softplus + i*li
    float lr = -softplus_f(lamRe[e]);
    float li = lamIm[e];
    float den = lr * lr + li * li;
    for (int t = 0; t < N_T; ++t) {
      float ed = expf(lr * dt[t]);
      float dr = ed * cosf(li * dt[t]);
      float di = ed * sinf(li * dt[t]);
      int f = (b * N_T + t) * 64 + e;
      decR[f] = dr;
      decI[f] = di;
      float ur = dr - 1.f, ui = di;
      forR[f] = (ur * lr + ui * li) / den;
      forI[f] = (ui * lr - ur * li) / den;
    }
  }
  __syncthreads();
  if (tid < N_B * N_D) {
    for (int t = 0; t < N_T; ++t) {
      float ar = 0.f, ai = 0.f, gg = bgate[e];
      for (int k = 0; k < N_D; ++k) {
        float fr = fshR[b][t][k], fi = fshI[b][t][k];
        float wr = WsR[k * 64 + e], wi = WsI[k * 64 + e];
        ar += fr * wr - fi * wi;
        ai += fr * wi + fi * wr;
        gg += fr * Wgate[k * 64 + e];
      }
      int f = (b * N_T + t) * 64 + e;
      srcR[f] = ar;
      srcI[f] = ai;
      gbuf[f] = 1.f / (1.f + expf(-gg));
    }
  }
}

// --------------------- 6) time recurrence ----------------------------------
// Reads f32 x_eigen, writes bf16 u (K-contiguous A operand for decode GEMM).

__global__ void scan_kernel(const float* __restrict__ xeR, const float* __restrict__ xeI,
                            bf16_t* __restrict__ utR, bf16_t* __restrict__ utI,
                            const float* __restrict__ srcR, const float* __restrict__ srcI,
                            const float* __restrict__ gbuf,
                            const float* __restrict__ decR, const float* __restrict__ decI,
                            const float* __restrict__ forR, const float* __restrict__ forI,
                            const float* __restrict__ hR, const float* __restrict__ hI) {
  int tid = blockIdx.x * blockDim.x + threadIdx.x;
  if (tid >= N_B * HW * N_D) return;
  int e = tid & 63;
  int rest = tid >> 6;
  int pix = rest % HW;
  int b = rest / HW;
  float sr = 0.f, si = 0.f;
  for (int t = 0; t < N_T; ++t) {
    int f = (b * N_T + t) * 64 + e;
    size_t xi = ((size_t)(b * N_T + t) * HW + pix) * 64 + e;
    float g = gbuf[f];
    float fr = xeR[xi] * g + srcR[f] * (1.f - g);
    float fi = xeI[xi] * g + srcI[f] * (1.f - g);
    float ofr = forR[f], ofi = forI[f];
    float ur = fr * ofr - fi * ofi;
    float ui = fr * ofi + fi * ofr;
    float dr = decR[f], di = decI[f];
    if (t == 0) {
      size_t hidx = ((size_t)b * HW + pix) * 64 + e;
      float hr = hR[hidx], hi = hI[hidx];
      ur += hr * dr - hi * di;
      ui += hr * di + hi * dr;
    }
    float nsr = dr * sr - di * si + ur;
    float nsi = dr * si + di * sr + ui;
    sr = nsr;
    si = nsi;
    utR[xi] = (bf16_t)sr;
    utI[xi] = (bf16_t)si;
  }
}

// -------------------- 8) LN2 + softmax router gates ------------------------
// Writes bf16 xn2 (pos*128, K-contiguous MoE A operand + final residual term).

__global__ void ln2_gates_kernel(const float* __restrict__ xdR, const float* __restrict__ xdI,
                                 const float* __restrict__ gtg, const float* __restrict__ gtb,
                                 const float* __restrict__ Wgm,
                                 bf16_t* __restrict__ xn2b, float* __restrict__ gates) {
  size_t p = (size_t)blockIdx.x * blockDim.x + threadIdx.x;
  if (p >= N_POS) return;
  const float* r = xdR + p * 64;
  const float* im = xdI + p * 64;
  float s = 0.f, ss = 0.f;
  for (int d = 0; d < 64; ++d) {
    float a = r[d], c = im[d];
    s += a + c;
    ss += a * a + c * c;
  }
  float mean = s * (1.f / 128.f);
  float var = ss * (1.f / 128.f) - mean * mean;
  float inv = rsqrtf(var + 1e-5f);
  bf16_t* o = xn2b + p * 128;
  float lg[4] = {0.f, 0.f, 0.f, 0.f};
  for (int d = 0; d < 64; ++d) {
    float vr = (r[d] - mean) * inv * gtg[d] + gtb[d];
    float vi = (im[d] - mean) * inv * gtg[64 + d] + gtb[64 + d];
    o[d] = (bf16_t)vr;
    o[64 + d] = (bf16_t)vi;
#pragma unroll
    for (int e2 = 0; e2 < 4; ++e2)
      lg[e2] += vr * Wgm[d * 4 + e2] + vi * Wgm[(64 + d) * 4 + e2];
  }
  float mx = fmaxf(fmaxf(lg[0], lg[1]), fmaxf(lg[2], lg[3]));
  float ex[4], se = 0.f;
#pragma unroll
  for (int e2 = 0; e2 < 4; ++e2) {
    ex[e2] = expf(lg[e2] - mx);
    se += ex[e2];
  }
#pragma unroll
  for (int e2 = 0; e2 < 4; ++e2) gates[p * 4 + e2] = ex[e2] / se;
}

// --------------------------- 9) MoE with WMMA ------------------------------
// One wave = 16 positions.  A-fragments cached in registers; hdn staged bf16
// in wave-private LDS (b128 ds loads); pre-swizzled W1/W2 fragments.

__global__ void __launch_bounds__(128) moe_wmma_kernel(
    const bf16_t* __restrict__ xn2b, const float* __restrict__ gates,
    const bf16_t* __restrict__ fW1, const float* __restrict__ b1,
    const bf16_t* __restrict__ fW2, const float* __restrict__ b2,
    float* __restrict__ delta) {
  __shared__ __align__(16) bf16_t hdn_s[4][16 * N_F];
  __shared__ float g_s[4][16 * 4];
  int wslot = threadIdx.x >> 5;
  int lane = threadIdx.x & 31;
  int wid = blockIdx.x * 4 + wslot;
  if (wid >= N_POS / 16) return;
  size_t pos0 = (size_t)wid * 16;
  bf16_t* hdn = hdn_s[wslot];
  float* gl = g_s[wslot];
  if (lane < 16) {
#pragma unroll
    for (int e2 = 0; e2 < 4; ++e2) gl[lane * 4 + e2] = gates[(pos0 + lane) * 4 + e2];
  }
  int m = lane & 15, half = (lane >> 4) & 1;
  v16bf axn[4];
#pragma unroll
  for (int kc = 0; kc < 4; ++kc)
    axn[kc] = load_a_rm(xn2b + pos0 * 128, 128, m, kc * 32, lane);
  v8f dacc[8];
#pragma unroll
  for (int i = 0; i < 8; ++i) dacc[i] = v8f_zero();

  for (int e2 = 0; e2 < N_EXP; ++e2) {
    const bf16_t* f1 = fW1 + (size_t)e2 * 64 * FRAG_ELEMS;
    const bf16_t* f2 = fW2 + (size_t)e2 * 64 * FRAG_ELEMS;
    __builtin_prefetch(f1, 0, 1);
    __builtin_prefetch(f2, 0, 1);
    // hdn = gelu(xn2 @ W1[e] + b1[e])   (16 x 256)
    for (int nt = 0; nt < 16; ++nt) {
      v8f acc = v8f_zero();
#pragma unroll
      for (int kc = 0; kc < 4; ++kc) {
        v16bf bw = load_frag(f1 + (size_t)(nt * 4 + kc) * FRAG_ELEMS, lane);
        acc = wmma_bf16(axn[kc], bw, acc);
      }
      float bias = b1[e2 * N_F + nt * 16 + m];
#pragma unroll
      for (int v = 0; v < 8; ++v) {
        int row = v + 8 * half;
        hdn[row * N_F + nt * 16 + m] = (bf16_t)gelu_tanh(acc[v] + bias);
      }
    }
    // delta += gate_e * (hdn @ W2[e] + b2[e])   (16 x 128)
    for (int nt = 0; nt < 8; ++nt) {
      v8f acc = v8f_zero();
#pragma unroll
      for (int kc = 0; kc < 8; ++kc) {
        v16bf ah = load_a_rm(hdn, N_F, m, kc * 32, lane);
        v16bf bw = load_frag(f2 + (size_t)(nt * 8 + kc) * FRAG_ELEMS, lane);
        acc = wmma_bf16(ah, bw, acc);
      }
      float bias2 = b2[e2 * 128 + nt * 16 + m];
#pragma unroll
      for (int v = 0; v < 8; ++v) {
        int row = v + 8 * half;
        dacc[nt][v] += gl[row * 4 + e2] * (acc[v] + bias2);
      }
    }
  }
#pragma unroll
  for (int nt = 0; nt < 8; ++nt)
#pragma unroll
    for (int v = 0; v < 8; ++v)
      delta[(pos0 + v + 8 * half) * 128 + nt * 16 + m] = dacc[nt][v];
}

// --------------------- 10) final residual + transpose ----------------------

__global__ void final_kernel(const float* __restrict__ xpR, const float* __restrict__ xpI,
                             const bf16_t* __restrict__ xn2b, const float* __restrict__ delta,
                             float* __restrict__ out) {
  size_t i = (size_t)blockIdx.x * blockDim.x + threadIdx.x;
  const size_t TOT = (size_t)N_IMG * 128 * HW;
  if (i >= TOT) return;
  size_t img = i / ((size_t)128 * HW);
  int c = (int)((i / HW) % 128);
  int pix = (int)(i % HW);
  size_t pos = img * HW + pix;
  float v = (float)xn2b[pos * 128 + c] + delta[pos * 128 + c];
  if (c < 64)
    v += xpR[img * IMG_STRIDE + (size_t)c * HW + pix];
  else
    v += xpI[img * IMG_STRIDE + (size_t)(c - 64) * HW + pix];
  out[i] = v;
}

// ------------------------------ launcher -----------------------------------

extern "C" void kernel_launch(void* const* d_in, const int* in_sizes, int n_in,
                              void* d_out, int out_size, void* d_ws, size_t ws_size,
                              hipStream_t stream) {
  (void)in_sizes; (void)n_in; (void)out_size; (void)ws_size;
  const float* x_re    = (const float*)d_in[0];
  const float* x_im    = (const float*)d_in[1];
  const float* h_re    = (const float*)d_in[2];
  const float* h_im    = (const float*)d_in[3];
  const float* flux_re = (const float*)d_in[4];
  const float* flux_im = (const float*)d_in[5];
  const float* dt      = (const float*)d_in[6];
  const float* gs_g    = (const float*)d_in[7];
  const float* gs_b    = (const float*)d_in[8];
  const float* wc_re   = (const float*)d_in[9];
  const float* wc_im   = (const float*)d_in[10];
  const float* bc_re   = (const float*)d_in[11];
  const float* bc_im   = (const float*)d_in[12];
  const float* metric  = (const float*)d_in[13];
  const float* gt_g    = (const float*)d_in[14];
  const float* gt_b    = (const float*)d_in[15];
  const float* E_re    = (const float*)d_in[16];
  const float* E_im    = (const float*)d_in[17];
  const float* Einv_re = (const float*)d_in[18];
  const float* Einv_im = (const float*)d_in[19];
  const float* lam_flux= (const float*)d_in[20];
  const float* inj_re  = (const float*)d_in[21];
  const float* inj_im  = (const float*)d_in[22];
  const float* Wsrc_re = (const float*)d_in[23];
  const float* Wsrc_im = (const float*)d_in[24];
  const float* Wgate   = (const float*)d_in[25];
  const float* bgate   = (const float*)d_in[26];
  const float* lam_re  = (const float*)d_in[27];
  const float* lam_im  = (const float*)d_in[28];
  const float* Wg      = (const float*)d_in[29];
  const float* W1      = (const float*)d_in[30];
  const float* b1      = (const float*)d_in[31];
  const float* W2      = (const float*)d_in[32];
  const float* b2      = (const float*)d_in[33];

  // ---- workspace layout (lifetime-aliased), ~153 MB ----
  char* ws = (char*)d_ws;
  size_t off = 0;
  auto take = [&](size_t bytes) -> char* {
    char* p = ws + off;
    off += (bytes + 255) & ~(size_t)255;
    return p;
  };
  const size_t PLANE_F = (size_t)N_IMG * IMG_STRIDE * sizeof(float);  // 18,874,368 B
  const size_t HALF_PF = PLANE_F / 2;

  char* regionA = take(PLANE_F);            // nr|ni (bf16); reused as xdR (f32)
  bf16_t* nr = (bf16_t*)regionA;
  bf16_t* ni = (bf16_t*)(regionA + HALF_PF);
  float* xdR = (float*)regionA;             // alias: nr/ni dead after conv
  float* xpR = (float*)take(PLANE_F);       // post-conv x, f32 planar (re)
  float* xpI = (float*)take(PLANE_F);       // (im)
  char* regionT = take(PLANE_F);            // xtR|xtI (bf16); reused as xn2b
  bf16_t* xtR = (bf16_t*)regionT;
  bf16_t* xtI = (bf16_t*)(regionT + HALF_PF);
  bf16_t* xn2b = (bf16_t*)regionT;          // alias: xt dead after encode GEMM
  float* xeR = (float*)take(PLANE_F);       // x_eigen f32 (re)
  float* xeI = (float*)take(PLANE_F);       // (im), contiguous with xeR
  float* delta = xeR;                       // alias: xe dead after scan
  char* regionU = take(PLANE_F);            // utR|utI (bf16)
  bf16_t* utR = (bf16_t*)regionU;
  bf16_t* utI = (bf16_t*)(regionU + HALF_PF);
  float* xdI = (float*)take(PLANE_F);
  float* gates = (float*)take((size_t)N_POS * 4 * sizeof(float));
  const size_t FRAG_B = FRAG_ELEMS * sizeof(bf16_t);   // 1 KB / fragment
  bf16_t* fEr   = (bf16_t*)take(8 * FRAG_B);
  bf16_t* fEi   = (bf16_t*)take(8 * FRAG_B);
  bf16_t* fEin  = (bf16_t*)take(8 * FRAG_B);
  bf16_t* fVr   = (bf16_t*)take(8 * FRAG_B);
  bf16_t* fVi   = (bf16_t*)take(8 * FRAG_B);
  bf16_t* fVin  = (bf16_t*)take(8 * FRAG_B);
  bf16_t* fWr   = (bf16_t*)take(72 * FRAG_B);
  bf16_t* fWi   = (bf16_t*)take(72 * FRAG_B);
  bf16_t* fWin  = (bf16_t*)take(72 * FRAG_B);
  bf16_t* fW1   = (bf16_t*)take(256 * FRAG_B);
  bf16_t* fW2   = (bf16_t*)take(256 * FRAG_B);
  const size_t SM = (size_t)N_IMG * 64 * sizeof(float);  // 8 KB small arrays
  float* mR   = (float*)take(SM);
  float* mI   = (float*)take(SM);
  float* srcR = (float*)take(SM);
  float* srcI = (float*)take(SM);
  float* gbuf = (float*)take(SM);
  float* decR = (float*)take(SM);
  float* decI = (float*)take(SM);
  float* forR = (float*)take(SM);
  float* forI = (float*)take(SM);

  // ---- weight pre-swizzle (fragment-ordered bf16) ----
  swz_sq64_kernel<<<1, 256, 0, stream>>>(E_re, fEr, 1.f);
  swz_sq64_kernel<<<1, 256, 0, stream>>>(E_im, fEi, 1.f);
  swz_sq64_kernel<<<1, 256, 0, stream>>>(E_im, fEin, -1.f);
  swz_sq64_kernel<<<1, 256, 0, stream>>>(Einv_re, fVr, 1.f);
  swz_sq64_kernel<<<1, 256, 0, stream>>>(Einv_im, fVi, 1.f);
  swz_sq64_kernel<<<1, 256, 0, stream>>>(Einv_im, fVin, -1.f);
  swz_conv_kernel<<<9, 256, 0, stream>>>(wc_re, fWr, 1.f);
  swz_conv_kernel<<<9, 256, 0, stream>>>(wc_im, fWi, 1.f);
  swz_conv_kernel<<<9, 256, 0, stream>>>(wc_im, fWin, -1.f);
  swz_w1_kernel<<<32, 256, 0, stream>>>(W1, fW1);
  swz_w2_kernel<<<32, 256, 0, stream>>>(W2, fW2);

  // ---- pipeline ----
  ln1_kernel<<<(N_POS + 255) / 256, 256, 0, stream>>>(x_re, x_im, gs_g, gs_b, nr, ni);

  conv_wmma_kernel<<<(N_IMG * N_H * 3 * 4) / 4, 128, 0, stream>>>(
      nr, ni, fWr, fWi, fWin, bc_re, bc_im, metric, x_re, x_im, xpR, xpI, xtR, xtI);

  cgemm64_kernel<<<((N_POS / 16) * 4) / 4, 128, 0, stream>>>(
      xtR, xtI, fEr, fEi, fEin, xeR, xeI);  // eigen encode

  mean_kernel<<<N_IMG, 64, 0, stream>>>(xeR, xeI, mR, mI);

  flux_kernel<<<1, 128, 0, stream>>>(mR, mI, dt, lam_flux, inj_re, inj_im,
                                     flux_re, flux_im, Wsrc_re, Wsrc_im, Wgate, bgate,
                                     lam_re, lam_im, srcR, srcI, gbuf,
                                     decR, decI, forR, forI);

  scan_kernel<<<(N_B * HW * N_D) / 256, 256, 0, stream>>>(
      xeR, xeI, utR, utI, srcR, srcI, gbuf, decR, decI, forR, forI, h_re, h_im);

  cgemm64_kernel<<<((N_POS / 16) * 4) / 4, 128, 0, stream>>>(
      utR, utI, fVr, fVi, fVin, xdR, xdI);  // eigen decode

  ln2_gates_kernel<<<(N_POS + 255) / 256, 256, 0, stream>>>(
      xdR, xdI, gt_g, gt_b, Wg, xn2b, gates);

  moe_wmma_kernel<<<(N_POS / 16) / 4, 128, 0, stream>>>(
      xn2b, gates, fW1, b1, fW2, b2, delta);

  final_kernel<<<((size_t)N_IMG * 128 * HW) / 256, 256, 0, stream>>>(
      xpR, xpI, xn2b, delta, (float*)d_out);
}
